// CenterNet_30648886624869
// MI455X (gfx1250) — matrix operand order repaired and checked by reading, not compiled, not run on hardware
//
#include <hip/hip_runtime.h>

#define B_    16
#define H_    128
#define W_    128
#define C_    64
#define CLS   80
#define TOPK_ 40
#define HW_   (H_ * W_)

typedef __attribute__((ext_vector_type(16))) __bf16 v16bf;
typedef __attribute__((ext_vector_type(8)))  __bf16 v8bf;
typedef __attribute__((ext_vector_type(8)))  float  v8f;

union ABf { v16bf v; v8bf h[2]; };

__device__ __forceinline__ __bf16 f2bf(float f) {
  unsigned u = __float_as_uint(f);
  u = (u + 0x7FFFu + ((u >> 16) & 1u)) >> 16;        // round-to-nearest-even
  unsigned short s = (unsigned short)u;
  return __builtin_bit_cast(__bf16, s);
}

// ---------------------------------------------------------------------------
// Kernel 1: repack weights/biases.
//   wp1: [192][576] bf16, n = head*64+co, k = tap*64+ci  (lane-contiguous K)
//   wp2: [96][192]  bf16, block-diagonal over hidden: n<80 hm (k 0..63),
//        n 80..81 wh (k 64..127), n 82..83 reg (k 128..191), rest zero
// ---------------------------------------------------------------------------
__global__ void pack_params_kernel(
    const float* __restrict__ hm_w1, const float* __restrict__ hm_b1,
    const float* __restrict__ hm_w2, const float* __restrict__ hm_b2,
    const float* __restrict__ wh_w1, const float* __restrict__ wh_b1,
    const float* __restrict__ wh_w2, const float* __restrict__ wh_b2,
    const float* __restrict__ reg_w1, const float* __restrict__ reg_b1,
    const float* __restrict__ reg_w2, const float* __restrict__ reg_b2,
    __bf16* __restrict__ wp1, __bf16* __restrict__ wp2,
    float* __restrict__ b1, float* __restrict__ b2)
{
  const int N1 = 192 * 576;
  const int N2 = 96 * 192;
  int i = blockIdx.x * blockDim.x + threadIdx.x;
  if (i < N1) {
    int n = i / 576, k = i % 576;
    int head = n >> 6, co = n & 63;
    const float* w = (head == 0) ? hm_w1 : (head == 1) ? wh_w1 : reg_w1;
    // HWIO: ((tap)*64 + ci)*64 + co == k*64 + co
    wp1[i] = f2bf(w[k * 64 + co]);
  } else if (i < N1 + N2) {
    int j = i - N1;
    int n = j / 192, k = j % 192;
    float v = 0.0f;
    if (n < 80)        { if (k < 64)              v = hm_w2[k * 80 + n]; }
    else if (n < 82)   { if (k >= 64 && k < 128)  v = wh_w2[(k - 64) * 2 + (n - 80)]; }
    else if (n < 84)   { if (k >= 128)            v = reg_w2[(k - 128) * 2 + (n - 82)]; }
    wp2[j] = f2bf(v);
  } else if (i < N1 + N2 + 192) {
    int n = i - N1 - N2;
    int head = n >> 6, co = n & 63;
    b1[n] = (head == 0) ? hm_b1[co] : (head == 1) ? wh_b1[co] : reg_b1[co];
  } else if (i < N1 + N2 + 192 + 96) {
    int n = i - N1 - N2 - 192;
    float v = 0.0f;
    if (n < 80)      v = hm_b2[n];
    else if (n < 82) v = wh_b2[n - 80];
    else if (n < 84) v = reg_b2[n - 82];
    b2[n] = v;
  }
}

// ---------------------------------------------------------------------------
// Kernel 2: fused conv3x3(+ReLU) -> conv1x1 for all 3 heads via bf16 WMMA.
// Block = 256 threads = 8 waves; tile = 64 spatial (one row segment) x 192 ch.
// ---------------------------------------------------------------------------
__global__ __launch_bounds__(256)
void conv_heads_kernel(const float* __restrict__ xin,
                       const __bf16* __restrict__ wp1,
                       const __bf16* __restrict__ wp2,
                       const float* __restrict__ b1,
                       const float* __restrict__ b2,
                       float* __restrict__ heat_t,
                       float* __restrict__ whreg)
{
  __shared__ __align__(16) __bf16 xs[3 * 66 * 64];   // rows y-1..y+1, cols x0-1..x0+64
  __shared__ __align__(16) __bf16 hid[64 * 192];     // hidden tile after ReLU

  const unsigned tid = threadIdx.x;
  const unsigned blk = blockIdx.x;        // B * H * (W/64) = 4096 blocks
  const int b   = (int)(blk >> 8);        // 256 blocks per batch
  const int rem = (int)(blk & 255u);
  const int y0  = rem >> 1;
  const int x0  = (rem & 1) * 64;

  __builtin_prefetch(wp1 + (size_t)tid * 432, 0, 0);   // warm weights in L2

  // ---- Phase 1: stage input row band into LDS as bf16 (SAME zero pad) ----
  for (unsigned i = tid; i < 3u * 66u * 16u; i += 256u) {
    unsigned r   = i / 1056u;             // 66*16
    unsigned q   = i - r * 1056u;
    unsigned col = q >> 4;
    unsigned c4  = (q & 15u) * 4u;
    int y  = y0 + (int)r - 1;
    int gx = x0 + (int)col - 1;
    float4 v = make_float4(0.f, 0.f, 0.f, 0.f);
    if ((unsigned)y < 128u && (unsigned)gx < 128u)
      v = *(const float4*)(xin + ((((size_t)b * 128 + y) * 128 + gx) * 64 + c4));
    __bf16* d = &xs[(r * 66u + col) * 64u + c4];
    d[0] = f2bf(v.x); d[1] = f2bf(v.y); d[2] = f2bf(v.z); d[3] = f2bf(v.w);
  }
  __syncthreads();

  // ---- Phase 2: conv3x3 GEMM, K = 576 (9 taps x 64ch), N-half per wave ----
  const unsigned wv    = tid >> 5;
  const unsigned lane  = tid & 31u;
  const unsigned lhalf = (lane < 16u) ? 0u : 1u;
  const unsigned lidx  = lane & 15u;
  const unsigned mg = wv & 3u;            // spatial sub-tile 0..3
  const unsigned ng = wv >> 2;            // channel half 0..1
  const unsigned m0 = mg * 16u;

  v8f acc[6];
#pragma unroll
  for (int nt = 0; nt < 6; nt++) {
    float bv = b1[ng * 96u + (unsigned)nt * 16u + lidx];
    acc[nt] = (v8f){bv, bv, bv, bv, bv, bv, bv, bv};
  }

  for (int tap = 0; tap < 9; tap++) {
    const int dy = tap / 3, dx = tap % 3;
#pragma unroll
    for (int ks = 0; ks < 2; ks++) {
      const unsigned c0 = (unsigned)ks * 32u;
      const unsigned cb = c0 + lhalf * 16u;
      const v8bf* ap = (const v8bf*)&xs[((unsigned)dy * 66u + (m0 + lidx + (unsigned)dx)) * 64u + cb];
      ABf a; a.h[0] = ap[0]; a.h[1] = ap[1];          // K {0..15} (+16 for upper lane half)
      const unsigned koff = (unsigned)tap * 64u + c0 + lhalf * 16u;
#pragma unroll
      for (int nt = 0; nt < 6; nt++) {
        const unsigned n = ng * 96u + (unsigned)nt * 16u + lidx;
        v16bf bfr = *(const v16bf*)(wp1 + (size_t)n * 576 + koff);
        acc[nt] = __builtin_amdgcn_wmma_f32_16x16x32_bf16(
            false, a.v, false, bfr, (short)0, acc[nt], false, false);
      }
    }
  }

  // ReLU -> bf16 hidden tile in LDS
#pragma unroll
  for (int nt = 0; nt < 6; nt++) {
    const unsigned ch = ng * 96u + (unsigned)nt * 16u + lidx;
#pragma unroll
    for (int r = 0; r < 8; r++) {
      const unsigned M = m0 + lhalf * 8u + (unsigned)r;
      hid[M * 192u + ch] = f2bf(fmaxf(acc[nt][r], 0.0f));
    }
  }
  __syncthreads();

  // ---- Phase 3: fused 1x1 convs: [64 x 192] x [192 x 96] block-diagonal ----
  for (int t = 0; t < 3; t++) {
    const unsigned id  = wv * 3u + (unsigned)t;       // 0..23 tiles
    const unsigned mg2 = id / 6u;
    const unsigned nt2 = id % 6u;
    const unsigned n   = nt2 * 16u + lidx;
    float bv = b2[n];
    v8f acc2 = (v8f){bv, bv, bv, bv, bv, bv, bv, bv};
#pragma unroll
    for (int kc = 0; kc < 6; kc++) {
      const unsigned c0 = (unsigned)kc * 32u;
      const unsigned cb = c0 + lhalf * 16u;
      const unsigned s  = mg2 * 16u + lidx;
      const v8bf* ap = (const v8bf*)&hid[s * 192u + cb];
      ABf a; a.h[0] = ap[0]; a.h[1] = ap[1];
      const unsigned koff = c0 + lhalf * 16u;
      v16bf bfr = *(const v16bf*)(wp2 + (size_t)n * 192 + koff);
      acc2 = __builtin_amdgcn_wmma_f32_16x16x32_bf16(
          false, a.v, false, bfr, (short)0, acc2, false, false);
    }
#pragma unroll
    for (int r = 0; r < 8; r++) {
      const unsigned s = mg2 * 16u + lhalf * 8u + (unsigned)r;
      const int x = x0 + (int)s;
      const float v = acc2[r];
      if (n < 80u) {
        // heat in [B][C][H][W] layout (class-major flat index for top-k)
        heat_t[(((size_t)b * CLS + n) * H_ + y0) * W_ + x] = 1.0f / (1.0f + __expf(-v));
      } else if (n < 84u) {
        // [B][HW][4] = {wh0, wh1, reg0, reg1}
        whreg[(((size_t)b * H_ + y0) * W_ + x) * 4 + (n - 80u)] = v;
      }
    }
  }
}

// ---------------------------------------------------------------------------
// Kernel 3: 3x3 peak mask (NMS) computed on the fly + local top-40 per chunk.
// Grid = B * 160 chunks of 8192 elements of the class-major heat map.
// ---------------------------------------------------------------------------
__global__ __launch_bounds__(256)
void nms_topk_stage1(const float* __restrict__ heat_t,
                     float* __restrict__ cand_v, int* __restrict__ cand_i)
{
  __shared__ float sv[8192];
  __shared__ float rv[256];
  __shared__ int   ri[256];
  const unsigned tid = threadIdx.x;
  const int b     = (int)(blockIdx.x / 160);
  const int chunk = (int)(blockIdx.x % 160);
  const int base  = chunk * 8192;
  const float* hb = heat_t + (size_t)b * (CLS * HW_);

  for (unsigned i = tid; i < 8192u; i += 256u) {
    int f = base + (int)i;
    int c = f >> 14;
    int p = f & 16383;
    int y = p >> 7;
    int x = p & 127;
    float v = hb[f];
    float m = v;
    const float* hc = hb + (size_t)c * HW_;
    for (int dy = -1; dy <= 1; dy++)
      for (int dx = -1; dx <= 1; dx++) {
        int yy = y + dy, xx = x + dx;
        if ((unsigned)yy < 128u && (unsigned)xx < 128u)
          m = fmaxf(m, hc[yy * 128 + xx]);
      }
    sv[i] = (v >= m) ? v : 0.0f;   // heat * (pooled == heat)
  }
  __syncthreads();

  for (int r = 0; r < TOPK_; r++) {
    float bv = -1e30f; int bi = 0;
    for (unsigned i = tid; i < 8192u; i += 256u) {
      float v = sv[i];
      if (v > bv) { bv = v; bi = (int)i; }
    }
    rv[tid] = bv; ri[tid] = bi;
    __syncthreads();
    for (int off = 128; off > 0; off >>= 1) {
      if (tid < (unsigned)off) {
        float ov = rv[tid + off]; int oi = ri[tid + off];
        if (ov > rv[tid] || (ov == rv[tid] && oi < ri[tid])) { rv[tid] = ov; ri[tid] = oi; }
      }
      __syncthreads();
    }
    if (tid == 0) {
      cand_v[((size_t)b * 160 + chunk) * TOPK_ + r] = rv[0];
      cand_i[((size_t)b * 160 + chunk) * TOPK_ + r] = base + ri[0];
      sv[ri[0]] = -1e30f;
    }
    __syncthreads();
  }
}

// ---------------------------------------------------------------------------
// Kernel 4: merge 160*40 candidates per batch into global top-40 (descending).
// ---------------------------------------------------------------------------
__global__ __launch_bounds__(256)
void topk_stage2(const float* __restrict__ cand_v, const int* __restrict__ cand_i,
                 float* __restrict__ scores, int* __restrict__ idxs)
{
  __shared__ float sv[6400];
  __shared__ int   si[6400];
  __shared__ float rv[256];
  __shared__ int   ri[256];
  __shared__ int   rs[256];
  const unsigned tid = threadIdx.x;
  const int b = (int)blockIdx.x;
  for (unsigned i = tid; i < 6400u; i += 256u) {
    sv[i] = cand_v[(size_t)b * 6400 + i];
    si[i] = cand_i[(size_t)b * 6400 + i];
  }
  __syncthreads();
  for (int r = 0; r < TOPK_; r++) {
    float bv = -1e30f; int bi = 0x7fffffff; int bs = 0;
    for (unsigned i = tid; i < 6400u; i += 256u) {
      float v = sv[i]; int gi = si[i];
      if (v > bv || (v == bv && gi < bi)) { bv = v; bi = gi; bs = (int)i; }
    }
    rv[tid] = bv; ri[tid] = bi; rs[tid] = bs;
    __syncthreads();
    for (int off = 128; off > 0; off >>= 1) {
      if (tid < (unsigned)off) {
        float ov = rv[tid + off]; int oi = ri[tid + off]; int os = rs[tid + off];
        if (ov > rv[tid] || (ov == rv[tid] && oi < ri[tid])) {
          rv[tid] = ov; ri[tid] = oi; rs[tid] = os;
        }
      }
      __syncthreads();
    }
    if (tid == 0) {
      scores[b * TOPK_ + r] = rv[0];
      idxs[b * TOPK_ + r]   = ri[0];
      sv[rs[0]] = -1e30f;
    }
    __syncthreads();
  }
}

// ---------------------------------------------------------------------------
// Kernel 5: gather wh/reg at top-k, build boxes (ascending-score order, *4).
// ---------------------------------------------------------------------------
__global__ void gather_bbox(const float* __restrict__ scores, const int* __restrict__ idxs,
                            const float* __restrict__ whreg, float* __restrict__ out)
{
  int t = blockIdx.x * blockDim.x + threadIdx.x;
  if (t >= B_ * TOPK_) return;
  int b = t / TOPK_, r = t % TOPK_;
  int f = idxs[b * TOPK_ + r];
  float sc = scores[b * TOPK_ + r];
  int c = f >> 14;
  int p = f & 16383;
  float ys = (float)(p >> 7);
  float xs = (float)(p & 127);
  const float4 wr = *(const float4*)(whreg + ((size_t)b * HW_ + p) * 4);
  float w = wr.x, h = wr.y, ox = wr.z, oy = wr.w;
  float cx = xs + ox, cy = ys + oy;
  int kk = TOPK_ - 1 - r;                 // reverse: reference returns ascending
  float* o = out + ((size_t)b * TOPK_ + kk) * 6;
  o[0] = (cx - 0.5f * w) * 4.0f;
  o[1] = (cy - 0.5f * h) * 4.0f;
  o[2] = (cx + 0.5f * w) * 4.0f;
  o[3] = (cy + 0.5f * h) * 4.0f;
  o[4] = (float)c;
  o[5] = sc;
}

// ---------------------------------------------------------------------------
extern "C" void kernel_launch(void* const* d_in, const int* in_sizes, int n_in,
                              void* d_out, int out_size, void* d_ws, size_t ws_size,
                              hipStream_t stream) {
  (void)in_sizes; (void)n_in; (void)out_size; (void)ws_size;
  const float* xin    = (const float*)d_in[0];
  const float* hm_w1  = (const float*)d_in[1];
  const float* hm_b1  = (const float*)d_in[2];
  const float* hm_w2  = (const float*)d_in[3];
  const float* hm_b2  = (const float*)d_in[4];
  const float* wh_w1  = (const float*)d_in[5];
  const float* wh_b1  = (const float*)d_in[6];
  const float* wh_w2  = (const float*)d_in[7];
  const float* wh_b2  = (const float*)d_in[8];
  const float* reg_w1 = (const float*)d_in[9];
  const float* reg_b1 = (const float*)d_in[10];
  const float* reg_w2 = (const float*)d_in[11];
  const float* reg_b2 = (const float*)d_in[12];
  float* out = (float*)d_out;

  char* ws = (char*)d_ws;
  size_t o = 0;
  auto take = [&](size_t bytes) -> char* {
    o = (o + 255) & ~(size_t)255;
    char* p = ws + o;
    o += bytes;
    return p;
  };
  __bf16* wp1     = (__bf16*)take((size_t)192 * 576 * 2);
  __bf16* wp2     = (__bf16*)take((size_t)96 * 192 * 2);
  float*  b1      = (float*)take((size_t)192 * 4);
  float*  b2      = (float*)take((size_t)96 * 4);
  float*  heat_t  = (float*)take((size_t)B_ * CLS * HW_ * 4);   // 80 MiB
  float*  whreg   = (float*)take((size_t)B_ * HW_ * 4 * 4);     // 4 MiB
  float*  cand_v  = (float*)take((size_t)B_ * 160 * TOPK_ * 4);
  int*    cand_i  = (int*)take((size_t)B_ * 160 * TOPK_ * 4);
  float*  scores  = (float*)take((size_t)B_ * TOPK_ * 4);
  int*    idxs    = (int*)take((size_t)B_ * TOPK_ * 4);

  // 1) repack params to bf16 WMMA layout
  {
    int total = 192 * 576 + 96 * 192 + 192 + 96;
    pack_params_kernel<<<(total + 255) / 256, 256, 0, stream>>>(
        hm_w1, hm_b1, hm_w2, hm_b2, wh_w1, wh_b1, wh_w2, wh_b2,
        reg_w1, reg_b1, reg_w2, reg_b2, wp1, wp2, b1, b2);
  }
  // 2) fused conv3x3 + ReLU + conv1x1 for all 3 heads (WMMA)
  conv_heads_kernel<<<B_ * H_ * (W_ / 64), 256, 0, stream>>>(
      xin, wp1, wp2, b1, b2, heat_t, whreg);
  // 3) NMS mask + per-chunk top-40
  nms_topk_stage1<<<B_ * 160, 256, 0, stream>>>(heat_t, cand_v, cand_i);
  // 4) merge to global top-40 per batch
  topk_stage2<<<B_, 256, 0, stream>>>(cand_v, cand_i, scores, idxs);
  // 5) gather + box decode
  gather_bbox<<<(B_ * TOPK_ + 255) / 256, 256, 0, stream>>>(scores, idxs, whreg, out);
}
